// NGPRadianceField_37074157699561
// MI455X (gfx1250) — matrix operand order repaired and checked by reading, not compiled, not run on hardware
//
#include <hip/hip_runtime.h>

typedef __attribute__((ext_vector_type(16))) _Float16 v16h;
typedef __attribute__((ext_vector_type(8)))  _Float16 v8h;
typedef __attribute__((ext_vector_type(8)))  float    v8f;

#define NGP_LOG2_T 19
#define NGP_TABLE_SIZE (1u << NGP_LOG2_T)
#define NGP_TMASK (NGP_TABLE_SIZE - 1u)

// floor(16 * (4096/16)^(l/15)) for l = 0..15
__constant__ float c_res[16] = {
    16.f, 23.f, 33.f, 48.f, 70.f, 101.f, 147.f, 212.f,
    307.f, 445.f, 645.f, 933.f, 1350.f, 1955.f, 2829.f, 4096.f};

__device__ __forceinline__ v8f wmma32(v16h a, v16h b, v8f c) {
    return __builtin_amdgcn_wmma_f32_16x16x32_f16(
        false, a, false, b, (short)0, c, false, false);
}

// Single-instruction ReLU (select(x>0,x,0) == v_max_num(0,x) incl. NaN/-0 cases).
__device__ __forceinline__ float relu(float x) {
    float y;
    asm("v_max_num_f32_e32 %0, 0, %1" : "=v"(y) : "v"(x));
    return y;
}

// Load 16 contiguous halves (B-operand fragment for one lane).
__device__ __forceinline__ v16h ld16(const _Float16* p) {
    v8h lo = *(const v8h*)p;
    v8h hi = *(const v8h*)(p + 8);
    v16h r;
#pragma unroll
    for (int j = 0; j < 8; ++j) { r[j] = lo[j]; r[8 + j] = hi[j]; }
    return r;
}

// A-operand fragment from a row-major staged row:
// lane holds K = chunk*32 + hs*16 + {0..15}.
__device__ __forceinline__ v16h ldA(const _Float16* row, int chunk, int hs) {
    v8h lo = *(const v8h*)(row + chunk * 32 + hs * 16);
    v8h hi = *(const v8h*)(row + chunk * 32 + hs * 16 + 8);
    v16h r;
#pragma unroll
    for (int j = 0; j < 8; ++j) { r[j] = lo[j]; r[8 + j] = hi[j]; }
    return r;
}

// One hash-grid level, phase-split so all 8 corner gathers are in flight
// before any accumulation. Level is folded into a 32-bit element index so the
// compiler can use uniform-base + 32-bit voffset (GVS) addressing.
__device__ __forceinline__ void enc_level(int l, float x0, float x1, float x2,
                                          const float2* __restrict__ tbl,
                                          float& f0, float& f1) {
    const float r = c_res[l];
    float p0 = x0 * r, p1 = x1 * r, p2 = x2 * r;
    float fl0 = floorf(p0), fl1 = floorf(p1), fl2 = floorf(p2);
    unsigned i0 = (unsigned)(int)fl0, i1 = (unsigned)(int)fl1, i2 = (unsigned)(int)fl2;
    float w0 = p0 - fl0, w1 = p1 - fl1, w2 = p2 - fl2;

    // xor-delta hash: (i+1)*p == i*p + p, so no per-corner multiplies
    unsigned hx[2] = {i0, i0 + 1u};
    unsigned hy[2] = {i1 * 2654435761u, i1 * 2654435761u + 2654435761u};
    unsigned hz[2] = {i2 * 805459861u,  i2 * 805459861u + 805459861u};

    const unsigned lbase = (unsigned)l << NGP_LOG2_T;
    float2 f[8];
#pragma unroll
    for (int c = 0; c < 8; ++c) {            // phase 1: 8 gathers in flight
        unsigned h = (hx[(c >> 2) & 1] ^ hy[(c >> 1) & 1] ^ hz[c & 1]) & NGP_TMASK;
        f[c] = tbl[lbase + h];               // uniform base, 32-bit offset
    }
    float wx[2] = {1.f - w0, w0};
    float wy[2] = {1.f - w1, w1};
    float wz[2] = {1.f - w2, w2};
    float a0 = 0.f, a1 = 0.f;
#pragma unroll
    for (int c = 0; c < 8; ++c) {            // phase 2: weighted accumulate
        float w = wx[(c >> 2) & 1] * wy[(c >> 1) & 1] * wz[c & 1];
        a0 += w * f[c].x;
        a1 += w * f[c].y;
    }
    f0 = a0; f1 = a1;
}

__global__ void __launch_bounds__(256) ngp_fused(
    const float* __restrict__ positions, const float* __restrict__ aabb,
    const float* __restrict__ table,
    const float* __restrict__ w1, const float* __restrict__ b1,
    const float* __restrict__ w2, const float* __restrict__ b2,
    const float* __restrict__ hw1, const float* __restrict__ hb1,
    const float* __restrict__ hw2, const float* __restrict__ hb2,
    const float* __restrict__ hw3, const float* __restrict__ hb3,
    float* __restrict__ out, int n, int nTiles) {
    // Weights stored transposed ([out][in]) in f16 so each lane's B fragment
    // (16 consecutive K for its column) is a contiguous 32B LDS read.
    __shared__ __align__(128) _Float16 sW1[64][32];   // enc(32) -> 64
    __shared__ __align__(128) _Float16 sW2[16][64];   // 64 -> 16 (density+geo)
    __shared__ __align__(128) _Float16 sH1[64][32];   // emb(15, K-pad 32, row0 zero) -> 64
    __shared__ __align__(128) _Float16 sH2[64][64];   // 64 -> 64
    __shared__ __align__(128) _Float16 sH3[16][64];   // 64 -> 3 (N-pad 16)
    __shared__ float sB1[64], sB2[16], sHB1[64], sHB2[64], sHB3[16];
    __shared__ __align__(128) _Float16 stage[8][16][64];   // per-wave 16x64 tile
    __shared__ __align__(128) _Float16 stageE[8][16][32];  // per-wave base tile (K-pad)
    __shared__ float sSel[8][16];

    const int tid = threadIdx.x;
    for (int i = tid; i < 64 * 32; i += 256) { int o = i >> 5, k = i & 31; sW1[o][k] = (_Float16)w1[k * 64 + o]; }
    for (int i = tid; i < 16 * 64; i += 256) { int o = i >> 6, k = i & 63; sW2[o][k] = (_Float16)w2[k * 16 + o]; }
    for (int i = tid; i < 64 * 32; i += 256) {
        int o = i >> 5, k = i & 31;
        float v = (k >= 1 && k < 16) ? hw1[(k - 1) * 64 + o] : 0.f;  // row0 kills density col
        sH1[o][k] = (_Float16)v;
    }
    for (int i = tid; i < 64 * 64; i += 256) { int o = i >> 6, k = i & 63; sH2[o][k] = (_Float16)hw2[k * 64 + o]; }
    for (int i = tid; i < 16 * 64; i += 256) { int o = i >> 6, k = i & 63; sH3[o][k] = (_Float16)((o < 3) ? hw3[k * 3 + o] : 0.f); }
    if (tid < 64) { sB1[tid] = b1[tid]; sHB1[tid] = hb1[tid]; sHB2[tid] = hb2[tid]; }
    if (tid < 16) { sB2[tid] = b2[tid]; sHB3[tid] = (tid < 3) ? hb3[tid] : 0.f; }
    __syncthreads();

    const int wv = tid >> 5, lane = tid & 31;
    const int hs = lane >> 4;   // half-wave select
    const int mA = lane & 15;   // this lane's A-row / encode point
    const int nC = lane & 15;   // this lane's B/C column
    const float2* tbl = (const float2*)table;
    const float amin0 = aabb[0], amin1 = aabb[1], amin2 = aabb[2];
    const float inv0 = 1.f / (aabb[3] - amin0);
    const float inv1 = 1.f / (aabb[4] - amin1);
    const float inv2 = 1.f / (aabb[5] - amin2);

    // Zero the K-pad columns (16..31) of the base-tile staging once per wave.
    for (int i = lane; i < 16 * 16; i += 32) {
        int rr = i >> 4, cc = i & 15;
        stageE[wv][rr][16 + cc] = (_Float16)0.f;
    }

    // Hoist per-column bias scalars out of the tile loop.
    float biasL1[4], biasH1[4], biasH2[4];
#pragma unroll
    for (int t = 0; t < 4; ++t) {
        biasL1[t] = sB1[t * 16 + nC];
        biasH1[t] = sHB1[t * 16 + nC];
        biasH2[t] = sHB2[t * 16 + nC];
    }
    const float biasL2 = sB2[nC];
    const float biasH3 = sHB3[nC];

    const int waveGlobal = blockIdx.x * 8 + wv;
    const int waveCount = gridDim.x * 8;

    for (int tile = waveGlobal; tile < nTiles; tile += waveCount) {
        const int p = tile * 16 + mA;
        float x0 = (positions[p * 3 + 0] - amin0) * inv0;
        float x1 = (positions[p * 3 + 1] - amin1) * inv1;
        float x2 = (positions[p * 3 + 2] - amin2) * inv2;
        if (hs == 0)
            sSel[wv][mA] = (x0 > 0.f && x0 < 1.f && x1 > 0.f && x1 < 1.f &&
                            x2 > 0.f && x2 < 1.f) ? 1.f : 0.f;

        // --- Hash encode straight into the WMMA A layout.
        // Lane holds K = {hs*16+j}; feature k = 2*level+f, so the
        // lane's 8 levels are {hs*8+t} and {hs*8+4+t}.
        v16h a1;
#pragma unroll
        for (int t = 0; t < 4; ++t) {
            float f0, f1;
            enc_level(hs * 8 + t, x0, x1, x2, tbl, f0, f1);
            a1[2 * t] = (_Float16)f0; a1[2 * t + 1] = (_Float16)f1;
            enc_level(hs * 8 + 4 + t, x0, x1, x2, tbl, f0, f1);
            a1[8 + 2 * t] = (_Float16)f0; a1[8 + 2 * t + 1] = (_Float16)f1;
        }

        // --- L1: enc(32) @ w1(32x64) + b1, ReLU -> stage[16][64]
#pragma unroll
        for (int t = 0; t < 4; ++t) {
            v16h b = ld16(&sW1[t * 16 + nC][hs * 16]);
            v8f c;
#pragma unroll
            for (int r = 0; r < 8; ++r) c[r] = biasL1[t];
            c = wmma32(a1, b, c);
#pragma unroll
            for (int r = 0; r < 8; ++r)
                stage[wv][8 * hs + r][t * 16 + nC] = (_Float16)relu(c[r]);
        }
        asm volatile("s_wait_dscnt 0" ::: "memory");

        // --- L2: h(64) @ w2(64x16) + b2 -> base tile (col0 = density, 1..15 = emb)
        {
            v16h a0 = ldA(&stage[wv][mA][0], 0, hs);
            v16h a1k = ldA(&stage[wv][mA][0], 1, hs);
            v8f cb;
#pragma unroll
            for (int r = 0; r < 8; ++r) cb[r] = biasL2;
            cb = wmma32(a0, ld16(&sW2[nC][hs * 16]), cb);
            cb = wmma32(a1k, ld16(&sW2[nC][32 + hs * 16]), cb);

#pragma unroll
            for (int r = 0; r < 8; ++r)
                stageE[wv][8 * hs + r][nC] = (_Float16)cb[r];
            if (nC == 0) {  // density epilogue: exp(x-1) * selector
#pragma unroll
                for (int r = 0; r < 8; ++r) {
                    int row = 8 * hs + r;
                    out[(size_t)n * 3 + (size_t)tile * 16 + row] =
                        __expf(cb[r] - 1.f) * sSel[wv][row];
                }
            }
        }
        asm volatile("s_wait_dscnt 0" ::: "memory");

        // --- H1: emb(16, K-pad 32) @ hw1p + hb1, ReLU -> stage[16][64]
        {
            v16h aE = ldA(&stageE[wv][mA][0], 0, hs);
#pragma unroll
            for (int t = 0; t < 4; ++t) {
                v16h b = ld16(&sH1[t * 16 + nC][hs * 16]);
                v8f c;
#pragma unroll
                for (int r = 0; r < 8; ++r) c[r] = biasH1[t];
                c = wmma32(aE, b, c);
#pragma unroll
                for (int r = 0; r < 8; ++r)
                    stage[wv][8 * hs + r][t * 16 + nC] = (_Float16)relu(c[r]);
            }
        }
        asm volatile("s_wait_dscnt 0" ::: "memory");

        // --- H2: hh(64) @ hw2(64x64) + hb2, ReLU -> stage[16][64]
        {
            v16h a0 = ldA(&stage[wv][mA][0], 0, hs);
            v16h a1k = ldA(&stage[wv][mA][0], 1, hs);
            v8f cc[4];
#pragma unroll
            for (int t = 0; t < 4; ++t) {
                v8f c;
#pragma unroll
                for (int r = 0; r < 8; ++r) c[r] = biasH2[t];
                c = wmma32(a0, ld16(&sH2[t * 16 + nC][hs * 16]), c);
                c = wmma32(a1k, ld16(&sH2[t * 16 + nC][32 + hs * 16]), c);
                cc[t] = c;
            }
#pragma unroll
            for (int t = 0; t < 4; ++t)
#pragma unroll
                for (int r = 0; r < 8; ++r)
                    stage[wv][8 * hs + r][t * 16 + nC] = (_Float16)relu(cc[t][r]);
        }
        asm volatile("s_wait_dscnt 0" ::: "memory");

        // --- H3: hh(64) @ hw3p(64x16) + hb3, sigmoid -> rgb (cols 0..2)
        {
            v16h a0 = ldA(&stage[wv][mA][0], 0, hs);
            v16h a1k = ldA(&stage[wv][mA][0], 1, hs);
            v8f c;
#pragma unroll
            for (int r = 0; r < 8; ++r) c[r] = biasH3;
            c = wmma32(a0, ld16(&sH3[nC][hs * 16]), c);
            c = wmma32(a1k, ld16(&sH3[nC][32 + hs * 16]), c);
            if (nC < 3) {
#pragma unroll
                for (int r = 0; r < 8; ++r) {
                    int row = 8 * hs + r;
                    float s = 1.f / (1.f + __expf(-c[r]));
                    out[((size_t)tile * 16 + row) * 3 + nC] = s;
                }
            }
        }
    }
}

extern "C" void kernel_launch(void* const* d_in, const int* in_sizes, int n_in,
                              void* d_out, int out_size, void* d_ws, size_t ws_size,
                              hipStream_t stream) {
    (void)n_in; (void)out_size; (void)d_ws; (void)ws_size;
    const float* positions = (const float*)d_in[0];
    const float* aabb      = (const float*)d_in[1];
    const float* table     = (const float*)d_in[2];
    const float* w1  = (const float*)d_in[3];
    const float* b1  = (const float*)d_in[4];
    const float* w2  = (const float*)d_in[5];
    const float* b2  = (const float*)d_in[6];
    const float* hw1 = (const float*)d_in[7];
    const float* hb1 = (const float*)d_in[8];
    const float* hw2 = (const float*)d_in[9];
    const float* hb2 = (const float*)d_in[10];
    const float* hw3 = (const float*)d_in[11];
    const float* hb3 = (const float*)d_in[12];
    float* out = (float*)d_out;

    int n = in_sizes[0] / 3;
    if (n <= 0) return;
    int nTiles = n / 16;                      // N_PTS = 524288 -> 32768 tiles
    int blocks = (nTiles + 15) / 16;          // ~2 tiles per wave
    if (blocks < 1) blocks = 1;
    if (blocks > 8192) blocks = 8192;

    ngp_fused<<<blocks, 256, 0, stream>>>(positions, aabb, table,
                                          w1, b1, w2, b2,
                                          hw1, hb1, hw2, hb2, hw3, hb3,
                                          out, n, nTiles);
}
